// CrossModalContrastiveLoss_10282151706902
// MI455X (gfx1250) — compile-verified, matrix operand order checked by reading, not run on hardware
//
#include <hip/hip_runtime.h>
#include <hip/hip_bf16.h>
#include <stdint.h>

typedef __bf16 v16bf __attribute__((ext_vector_type(16)));
typedef float  v8f   __attribute__((ext_vector_type(8)));

#define INV_T 33.333333333333336f
#define BT  128          // block tile (rows x cols = BT x BT)
#define KT  32           // K slice per WMMA / per LDS tile
#define NID 2048         // patient id range per reference

// ---------------- fp32 -> bf16 (round-to-nearest-even) ----------------
__global__ void cvt_bf16_kernel(const float* __restrict__ in,
                                unsigned short* __restrict__ out, int n) {
  int i = blockIdx.x * blockDim.x + threadIdx.x;
  if (i < n) {
    unsigned u = __float_as_uint(in[i]);
    unsigned r = (u + 0x7FFFu + ((u >> 16) & 1u)) >> 16;
    out[i] = (unsigned short)r;
  }
}

// ---------------- patient id histogram ----------------
__global__ void hist_kernel(const int* __restrict__ ids,
                            int* __restrict__ hist, int B) {
  int i = blockIdx.x * blockDim.x + threadIdx.x;
  if (i < B) atomicAdd(&hist[ids[i] & (NID - 1)], 1);
}

// ---------------- Tensor Data Mover (TDM) tile load ----------------
#if defined(__has_builtin)
#if __has_builtin(__builtin_amdgcn_tensor_load_to_lds) && !defined(NO_TDM)
#define USE_TDM 1
#endif
#endif

#ifdef USE_TDM
typedef unsigned int u32x4v __attribute__((ext_vector_type(4)));
typedef int          i32x4v __attribute__((ext_vector_type(4)));
typedef int          i32x8v __attribute__((ext_vector_type(8)));

// Load a BT x KT tile of 2-byte elements starting at (row0, k) of a row-major
// (B x d) tensor into LDS at byte offset ldsOff, rows stored contiguously
// (row stride = KT elements = 64 bytes). D# built per CDNA5 ISA ch.8.
__device__ __forceinline__ void tdm_load_tile(const unsigned short* gbase,
                                              unsigned ldsOff,
                                              int row0, int k, int d, int B) {
  unsigned long long ga =
      (unsigned long long)(uintptr_t)(gbase + (size_t)row0 * d + k);
  u32x4v g0;
  g0.x = 0x1u;                                   // count=1, user mode
  g0.y = ldsOff;                                 // lds_addr (bytes)
  g0.z = (unsigned)(ga & 0xFFFFFFFFu);           // global_addr[31:0]
  g0.w = (unsigned)((ga >> 32) & 0x01FFFFFFu)    // global_addr[56:32]
       | (2u << 30);                             // type=2 ("image")
  i32x8v g1;
  g1[0] = (int)(1u << 16);                                        // data_size=2B, mask=0
  g1[1] = (int)(((unsigned)d & 0xFFFFu) << 16);                   // tensor_dim0[15:0]
  g1[2] = (int)((((unsigned)d >> 16) & 0xFFFFu)                   // tensor_dim0[31:16]
              | (((unsigned)B & 0xFFFFu) << 16));                 // tensor_dim1[15:0]
  g1[3] = (int)((((unsigned)B >> 16) & 0xFFFFu)                   // tensor_dim1[31:16]
              | ((unsigned)KT << 16));                            // tile_dim0 = KT
  g1[4] = (int)(unsigned)BT;                                      // tile_dim1 = BT, tile_dim2=0
  g1[5] = (int)(unsigned)d;                                       // tensor_dim0_stride[31:0]
  g1[6] = 0;                                                      // stride0[47:32] | stride1 lo
  g1[7] = 0;
  i32x4v gz4 = {0, 0, 0, 0};
  i32x8v gz8 = {0, 0, 0, 0, 0, 0, 0, 0};
  // clang-23 / amdgpu-toolchain 6-arg form: (g0, g1, g2, g3, x, cpol)
  __builtin_amdgcn_tensor_load_to_lds(g0, g1, gz4, gz4, gz8, 0);
}
#endif

// ---------------- fused sim GEMM + masked exp-sum rows/cols + diag ----------------
// grid = (B/BT, B/BT), block = 256 threads = 8 waves (wave32).
// Wave w computes rows [w*16, w*16+16) of the 128x128 tile across all 128 cols
// as 8 accumulators of v_wmma_f32_16x16x32_bf16. A/B K-slices are staged in
// LDS (double-buffered) by the Tensor Data Mover; compute overlaps the DMA.
// All fragment ds_loads are issued before the WMMA chain so the in-order DS
// counter lets WMMA nt overlap with loads nt+1..7 (partial s_wait_dscnt).
__global__ __launch_bounds__(256) void sim_gemm_kernel(
    const unsigned short* __restrict__ Abf,   // z_a bf16, row-major BxD
    const unsigned short* __restrict__ Bbf,   // z_t bf16, row-major BxD
    const int* __restrict__ ids,
    float* __restrict__ rowSum, float* __restrict__ colSum,
    float* __restrict__ pos, int B, int d) {
  __shared__ unsigned short shA[2][BT * KT];   // 8 KB each buffer
  __shared__ unsigned short shB[2][BT * KT];
  __shared__ float sRow[BT];
  __shared__ float sCol[BT];
  __shared__ int   sIdR[BT];
  __shared__ int   sIdC[BT];

  const int tid   = threadIdx.x;
  const int lane  = tid & 31;
  const int wave  = tid >> 5;           // 0..7
  const int gRow0 = blockIdx.y * BT;
  const int gCol0 = blockIdx.x * BT;

  if (tid < BT) {
    sRow[tid] = 0.0f; sCol[tid] = 0.0f;
    sIdR[tid] = ids[gRow0 + tid];
  } else {
    sIdC[tid - BT] = ids[gCol0 + (tid - BT)];
  }

  const int mW   = wave << 4;           // wave's row base inside tile
  const int half = lane >> 4;           // 0 | 1
  const int l16  = lane & 15;
  const int kSteps = d / KT;

  v8f acc[8];
  const v8f vzero = {0.f, 0.f, 0.f, 0.f, 0.f, 0.f, 0.f, 0.f};
#pragma unroll
  for (int nt = 0; nt < 8; ++nt) acc[nt] = vzero;

#ifdef USE_TDM
  // ---- prologue: DMA first K-slice into buffer 0 ----
  if (wave == 0) {
    tdm_load_tile(Abf, (unsigned)(uintptr_t)&shA[0][0], gRow0, 0, d, B);
    tdm_load_tile(Bbf, (unsigned)(uintptr_t)&shB[0][0], gCol0, 0, d, B);
    __builtin_amdgcn_s_wait_tensorcnt(0);
  }
  __syncthreads();
#else
  // ---- fallback: cooperative load of first K-slice ----
  {
    const int row = tid >> 1, ch = tid & 1;           // 2 threads per row, 16B each
    const uint4* sa = (const uint4*)(Abf + (size_t)(gRow0 + row) * d + ch * 16);
    const uint4* sb = (const uint4*)(Bbf + (size_t)(gCol0 + row) * d + ch * 16);
    uint4* da = (uint4*)&shA[0][row * KT + ch * 16];
    uint4* db = (uint4*)&shB[0][row * KT + ch * 16];
    da[0] = sa[0]; db[0] = sb[0];
  }
  __syncthreads();
#endif

  for (int kk = 0; kk < kSteps; ++kk) {
    const int b = kk & 1;

#ifdef USE_TDM
    if (kk + 1 < kSteps && wave == 0) {           // DMA next slice, overlapped
      const int kn = (kk + 1) * KT;
      tdm_load_tile(Abf, (unsigned)(uintptr_t)&shA[1 - b][0], gRow0, kn, d, B);
      tdm_load_tile(Bbf, (unsigned)(uintptr_t)&shB[1 - b][0], gCol0, kn, d, B);
    }
#endif

    // ---- batch all fragment loads from LDS (in-order DS counter) ----
    const unsigned short* aP = &shA[b][(mW + l16) * KT + (half << 3)];
    union { uint4 u[2]; v16bf v; } af;
    af.u[0] = *(const uint4*)(aP);        // K = +0..7 (half0) / +8..15 (half1)
    af.u[1] = *(const uint4*)(aP + 16);   // K = +16..23 / +24..31
    const v16bf aFrag = af.v;

    v16bf bFrag[8];
#pragma unroll
    for (int nt = 0; nt < 8; ++nt) {
      const unsigned short* bP = &shB[b][((nt << 4) + l16) * KT + (half << 3)];
      union { uint4 u[2]; v16bf v; } bfr;
      bfr.u[0] = *(const uint4*)(bP);
      bfr.u[1] = *(const uint4*)(bP + 16);
      bFrag[nt] = bfr.v;
    }

    // ---- 8 back-to-back WMMAs ----
#pragma unroll
    for (int nt = 0; nt < 8; ++nt) {
      acc[nt] = __builtin_amdgcn_wmma_f32_16x16x32_bf16(
          /*neg_a=*/false, aFrag, /*neg_b=*/false, bFrag[nt],
          /*c_mod=*/(short)0, acc[nt], /*reuse_a=*/false, /*reuse_b=*/false);
    }

#ifdef USE_TDM
    if (kk + 1 < kSteps && wave == 0) __builtin_amdgcn_s_wait_tensorcnt(0);
    __syncthreads();
#else
    if (kk + 1 < kSteps) {
      __syncthreads();                               // everyone done reading 1-b
      const int kn = (kk + 1) * KT;
      const int row = tid >> 1, ch = tid & 1;
      const uint4* sa = (const uint4*)(Abf + (size_t)(gRow0 + row) * d + kn + ch * 16);
      const uint4* sb = (const uint4*)(Bbf + (size_t)(gCol0 + row) * d + kn + ch * 16);
      uint4* da = (uint4*)&shA[1 - b][row * KT + ch * 16];
      uint4* db = (uint4*)&shB[1 - b][row * KT + ch * 16];
      da[0] = sa[0]; db[0] = sb[0];
    }
    __syncthreads();
#endif
  }

  // ---- epilogue: C/D layout -> lane(n = l16, half selects M+8), VGPR r = M%8 ----
  float rAcc[8];
#pragma unroll
  for (int r = 0; r < 8; ++r) rAcc[r] = 0.0f;

#pragma unroll
  for (int nt = 0; nt < 8; ++nt) {
    const int nLoc = (nt << 4) + l16;
    const int idc  = sIdC[nLoc];
    float colPart = 0.0f;
#pragma unroll
    for (int r = 0; r < 8; ++r) {
      const int mLoc = mW + r + (half << 3);
      float s = acc[nt][r] * INV_T;
      bool diag = (gRow0 + mLoc) == (gCol0 + nLoc);
      bool msk  = (sIdR[mLoc] != idc) || diag;
      float e = msk ? __expf(s) : 0.0f;
      colPart += e;
      rAcc[r] += e;
      if (diag) pos[gRow0 + mLoc] = s;   // positive logit (written exactly once)
    }
    colPart += __shfl_xor(colPart, 16, 32);          // fold the two M-halves
    if (half == 0) atomicAdd(&sCol[nLoc], colPart);  // ds_add_f32
  }

#pragma unroll
  for (int r = 0; r < 8; ++r) {                      // butterfly over 16 lanes/half
    float v = rAcc[r];
    v += __shfl_xor(v, 1, 32);
    v += __shfl_xor(v, 2, 32);
    v += __shfl_xor(v, 4, 32);
    v += __shfl_xor(v, 8, 32);
    if (l16 == 0) atomicAdd(&sRow[mW + r + (half << 3)], v);
  }
  __syncthreads();

  if (tid < BT) {
    atomicAdd(&rowSum[gRow0 + tid], sRow[tid]);
    atomicAdd(&colSum[gCol0 + tid], sCol[tid]);
  }
}

// ---------------- final reduction: loss scalar ----------------
__global__ void loss_kernel(const float* __restrict__ rowSum,
                            const float* __restrict__ colSum,
                            const float* __restrict__ pos,
                            const int* __restrict__ ids,
                            const int* __restrict__ hist,
                            float* __restrict__ out, int B) {
  __shared__ float sA[256];
  __shared__ float sB[256];
  __shared__ int   sC[256];
  float aR = 0.0f, aC = 0.0f;
  int cnt = 0;
  for (int i = threadIdx.x; i < B; i += blockDim.x) {
    if (hist[ids[i] & (NID - 1)] < B) {   // exists a cross-patient negative
      float p = pos[i];
      aR += __logf(rowSum[i]) - p;
      aC += __logf(colSum[i]) - p;
      cnt++;
    }
  }
  sA[threadIdx.x] = aR; sB[threadIdx.x] = aC; sC[threadIdx.x] = cnt;
  __syncthreads();
  for (int s = 128; s > 0; s >>= 1) {
    if ((int)threadIdx.x < s) {
      sA[threadIdx.x] += sA[threadIdx.x + s];
      sB[threadIdx.x] += sB[threadIdx.x + s];
      sC[threadIdx.x] += sC[threadIdx.x + s];
    }
    __syncthreads();
  }
  if (threadIdx.x == 0) {
    int c = sC[0];
    out[0] = (c > 0) ? (sA[0] + sB[0]) / (2.0f * (float)c) : 0.0f;
  }
}

extern "C" void kernel_launch(void* const* d_in, const int* in_sizes, int n_in,
                              void* d_out, int out_size, void* d_ws, size_t ws_size,
                              hipStream_t stream) {
  const float* z_a = (const float*)d_in[0];
  const float* z_t = (const float*)d_in[1];
  const int*   ids = (const int*)d_in[2];

  const int B = in_sizes[2];          // 8192
  const int d = in_sizes[0] / B;      // 256

  char* ws = (char*)d_ws;
  unsigned short* Abf = (unsigned short*)ws;  ws += (size_t)B * d * sizeof(unsigned short);
  unsigned short* Bbf = (unsigned short*)ws;  ws += (size_t)B * d * sizeof(unsigned short);
  float* rowSum = (float*)ws;                 ws += (size_t)B * sizeof(float);
  float* colSum = (float*)ws;                 ws += (size_t)B * sizeof(float);
  int*   hist   = (int*)ws;                   ws += (size_t)NID * sizeof(int);
  float* pos    = (float*)ws;                 ws += (size_t)B * sizeof(float);

  // zero rowSum | colSum | hist (contiguous) every call — capture-safe memset node
  (void)hipMemsetAsync(rowSum, 0,
                       (size_t)(2 * B) * sizeof(float) + (size_t)NID * sizeof(int), stream);

  const int n = B * d;
  cvt_bf16_kernel<<<(n + 255) / 256, 256, 0, stream>>>(z_a, Abf, n);
  cvt_bf16_kernel<<<(n + 255) / 256, 256, 0, stream>>>(z_t, Bbf, n);
  hist_kernel<<<(B + 255) / 256, 256, 0, stream>>>(ids, hist, B);

  dim3 grid(B / BT, B / BT);
  sim_gemm_kernel<<<grid, 256, 0, stream>>>(Abf, Bbf, ids, rowSum, colSum, pos, B, d);

  loss_kernel<<<1, 256, 0, stream>>>(rowSum, colSum, pos, ids, hist, (float*)d_out, B);
}
